// DeepBilateralNetCurves_76493367542179
// MI455X (gfx1250) — matrix operand order repaired and checked by reading, not compiled, not run on hardware
//
#include <hip/hip_runtime.h>
#include <hip/hip_bf16.h>

typedef __attribute__((ext_vector_type(16))) _Float16 v16h;
typedef __attribute__((ext_vector_type(8)))  _Float16 v8h;
typedef __attribute__((ext_vector_type(8)))  float    v8f;

union AFrag { v16h v; v8h h[2]; };

// ---------------------------------------------------------------------------
// Implicit-GEMM conv via v_wmma_f32_16x16x32_f16, K staged 64 at a time.
//  - M tile: 16 output channels (blockIdx.y)
//  - N tile: 64 output pixels (blockIdx.x), one 16-pixel sub-tile per wave
//  - 2 WMMAs per LDS staging round (f16 operands, f32 accum)
// Block = 128 threads = 4 wave32.  Wout is a power of two (shift passed).
// ---------------------------------------------------------------------------
template<int KS>
__global__ void conv_wmma_kernel(
    const float* __restrict__ in, const float* __restrict__ wgt,
    const float* __restrict__ bias, const float* __restrict__ bn_s,
    const float* __restrict__ bn_b, float* __restrict__ out,
    int Cin, int Cout, int Hin, int Win, int Hout, int WoutShift,
    int stride, int pad, int relu)
{
    constexpr int KK2 = KS * KS;
    __shared__ __align__(16) _Float16 As[16 * 64];   // [cout 16][k 64]
    __shared__ __align__(16) _Float16 Bs[64 * 64];   // [pixel 64][k 64]

    const int tid  = threadIdx.x;
    const int wave = tid >> 5;
    const int lane = tid & 31;
    const int half = lane >> 4;
    const int idx  = lane & 15;

    const int Wout      = 1 << WoutShift;
    const int wmask     = Wout - 1;
    const int b         = blockIdx.z;
    const int cout_base = blockIdx.y * 16;
    const int npix      = Hout << WoutShift;
    const int npix_base = blockIdx.x * 64;

    const int K       = Cin * KK2;
    const int rounds  = (K + 63) >> 6;

    v8f acc = {};

    for (int kc = 0; kc < rounds; ++kc) {
        const int kbase = kc << 6;

        // Stage weight tile A: [16 cout][64 k] (zero-padded)
        for (int t = tid; t < 16 * 64; t += 128) {
            const int m  = t >> 6;
            const int k  = kbase + (t & 63);
            const int co = cout_base + m;
            float v = 0.f;
            if (k < K && co < Cout) v = wgt[(size_t)co * K + k];
            As[t] = (_Float16)v;
        }
        // Stage im2col tile B: [64 pixel][64 k] (zero-padded)
        for (int t = tid; t < 64 * 64; t += 128) {
            const int n   = t >> 6;
            const int k   = kbase + (t & 63);
            const int pix = npix_base + n;
            float v = 0.f;
            if (k < K && pix < npix) {
                const int oy = pix >> WoutShift, ox = pix & wmask;
                const int ci = k / KK2;              // KK2 is 9 or 1: const div
                const int r  = k - ci * KK2;
                const int ky = r / KS, kx = r - ky * KS;
                const int iy = oy * stride + ky - pad;
                const int ix = ox * stride + kx - pad;
                if (iy >= 0 && iy < Hin && ix >= 0 && ix < Win)
                    v = in[(((size_t)b * Cin + ci) * Hin + iy) * Win + ix];
            }
            Bs[t] = (_Float16)v;
        }
        __syncthreads();

        // Two K=32 WMMA steps per staging round.
        #pragma unroll
        for (int c = 0; c < 2; ++c) {
            // A-frag: lane = M (mod 16); lanes 0-15 hold K {0..7,16..23},
            //         lanes 16-31 hold K {8..15,24..31}
            AFrag a, bm;
            const _Float16* ap = As + idx * 64 + c * 32 + half * 8;
            a.h[0] = *(const v8h*)(ap);
            a.h[1] = *(const v8h*)(ap + 16);
            // B-frag: lane = N (mod 16); lanes 0-15 hold K {0..7,16..23},
            //         lanes 16-31 hold K {8..15,24..31}
            const _Float16* bp = Bs + (wave * 16 + idx) * 64 + c * 32 + half * 8;
            bm.h[0] = *(const v8h*)(bp);
            bm.h[1] = *(const v8h*)(bp + 16);
            acc = __builtin_amdgcn_wmma_f32_16x16x32_f16(
                      false, a.v, false, bm.v, (short)0, acc, false, false);
        }
        __syncthreads();
    }

    // C/D layout: lane = N; VGPR r -> M = r (+8 for upper lane half)
    const int pix = npix_base + wave * 16 + idx;
    if (pix < npix) {
        const int oy = pix >> WoutShift, ox = pix & wmask;
        #pragma unroll
        for (int r = 0; r < 8; ++r) {
            const int co = cout_base + half * 8 + r;
            if (co < Cout) {
                float v = acc[r];
                if (bias) v += bias[co];
                if (bn_s) v = v * bn_s[co] + bn_b[co];
                if (relu) v = fmaxf(v, 0.f);
                out[(((size_t)b * Cout + co) * Hout + oy) << WoutShift | ox] = v;
            }
        }
    }
}

// ---------------------------------------------------------------------------
// Fully-connected, single block.  Input vector (2*In f32, contiguous) is
// broadcast to all threads -> stage it in LDS with async global->LDS b128
// copies (ASYNCcnt path), then each thread computes one output dot product.
// ---------------------------------------------------------------------------
__global__ void fc_kernel(const float* __restrict__ in, const float* __restrict__ w,
                          const float* __restrict__ bias, float* __restrict__ out,
                          int In, int Out, int relu)
{
    __shared__ __align__(16) float sbuf[2048];     // up to 2*1024 floats

    const int nbytes = 2 * In * (int)sizeof(float);
    const unsigned sbase = (unsigned)(size_t)(&sbuf[0]);   // low 32 bits = LDS offset
    for (int off = threadIdx.x * 16; off < nbytes; off += blockDim.x * 16) {
        const unsigned lds = sbase + (unsigned)off;
        const char* gp = (const char*)in + off;
        asm volatile("global_load_async_to_lds_b128 %0, %1, off"
                     :: "v"(lds), "v"(gp) : "memory");
    }
    asm volatile("s_wait_asynccnt 0x0" ::: "memory");
    __syncthreads();

    for (int i = threadIdx.x; i < 2 * Out; i += blockDim.x) {
        const int b = i / Out, o = i - b * Out;
        const float* ip = sbuf + b * In;
        const float* wp = w + (size_t)o * In;
        float acc = bias[o];
        for (int k = 0; k < In; ++k) acc += ip[k] * wp[k];
        if (relu) acc = fmaxf(acc, 0.f);
        out[i] = acc;
    }
}

// ---------------------------------------------------------------------------
// fusion = relu(global[b][c] + local[b][c][y][x])
// ---------------------------------------------------------------------------
__global__ void fuse_kernel(const float* __restrict__ l, const float* __restrict__ g,
                            float* __restrict__ out, int C, int HW, int total)
{
    const int i = blockIdx.x * blockDim.x + threadIdx.x;
    if (i >= total) return;
    const int bc = i / HW;               // b*C + c
    out[i] = fmaxf(l[i] + g[bc], 0.f);
}

// ---------------------------------------------------------------------------
// Fused guide-map + trilinear slice + affine apply at full resolution.
// 4 pixels per thread with b128 plane loads/stores (memory-bound kernel:
// ~50 MB total traffic -> ~2.2 us at 23.3 TB/s is the roofline).
// coeffs channel layout: ch = z*12 + nc, nc = o*4 + i; grid is cache-resident.
// ---------------------------------------------------------------------------
__global__ void guide_slice_kernel(
    const float* __restrict__ img, const float* __restrict__ coeffs,
    const float* __restrict__ ccm_w, const float* __restrict__ ccm_b,
    const float* __restrict__ shifts, const float* __restrict__ slopes,
    const float* __restrict__ proj_w, const float* __restrict__ proj_b,
    float* __restrict__ out, int H, int W)
{
    const int HW = H * W;
    const int q  = blockIdx.x * blockDim.x + threadIdx.x;
    if (q >= (HW >> 2)) return;
    const int b    = blockIdx.z;
    const int pix0 = q << 2;
    const int y    = pix0 / W;
    const int x0   = pix0 - y * W;      // W % 4 == 0 -> quad stays in one row

    const float* cb = coeffs + (size_t)b * 96 * 256;
    __builtin_prefetch(cb, 0, 3);       // global_prefetch_b8: warm coeff grid

    const size_t base = (size_t)b * 3 * HW + pix0;
    const float4 c0 = *(const float4*)(img + base);
    const float4 c1 = *(const float4*)(img + base + HW);
    const float4 c2 = *(const float4*)(img + base + 2 * (size_t)HW);
    const float xin[3][4] = {{c0.x, c0.y, c0.z, c0.w},
                             {c1.x, c1.y, c1.z, c1.w},
                             {c2.x, c2.y, c2.z, c2.w}};
    float res[3][4];

    const int L = 8, gh = 16, gw = 16;
    const float gy = (y + 0.5f) * gh / H;
    const float fy = floorf(gy - 0.5f);

    for (int s = 0; s < 4; ++s) {
        // ---- guide: ccm -> 16-pt piecewise-linear curve -> proj -> clip ----
        float gacc = proj_b[0];
        #pragma unroll
        for (int c = 0; c < 3; ++c) {
            float g = ccm_b[c];
            #pragma unroll
            for (int i = 0; i < 3; ++i) g += ccm_w[c * 3 + i] * xin[i][s];
            float cur = 0.f;
            #pragma unroll
            for (int p = 0; p < 16; ++p)
                cur += slopes[c * 16 + p] * fmaxf(g - shifts[c * 16 + p], 0.f);
            gacc += proj_w[c] * cur;
        }
        const float gd = fminf(fmaxf(gacc, 0.f), 1.f);

        // ---- trilinear slice of 8x16x16 grid, 12 channels ----
        const int   x  = x0 + s;
        const float gx = (x + 0.5f) * gw / W;
        const float gz = gd * L;
        const float fx = floorf(gx - 0.5f);
        const float fz = floorf(gz - 0.5f);

        float A[12];
        #pragma unroll
        for (int nc = 0; nc < 12; ++nc) A[nc] = 0.f;

        #pragma unroll
        for (int dz = 0; dz < 2; ++dz) {
            const int   zz = min(max((int)(fz + dz), 0), L - 1);
            const float wz = fmaxf(1.f - fabsf(fz + dz + 0.5f - gz), 0.f);
            #pragma unroll
            for (int dy = 0; dy < 2; ++dy) {
                const int   yy = min(max((int)(fy + dy), 0), gh - 1);
                const float wy = fmaxf(1.f - fabsf(fy + dy + 0.5f - gy), 0.f);
                #pragma unroll
                for (int dx = 0; dx < 2; ++dx) {
                    const int   xx = min(max((int)(fx + dx), 0), gw - 1);
                    const float wx = fmaxf(1.f - fabsf(fx + dx + 0.5f - gx), 0.f);
                    const float wgt = wz * wy * wx;
                    const float* cp = cb + (size_t)zz * 12 * 256 + yy * gw + xx;
                    #pragma unroll
                    for (int nc = 0; nc < 12; ++nc)
                        A[nc] += wgt * cp[nc * 256];
                }
            }
        }
        #pragma unroll
        for (int o = 0; o < 3; ++o)
            res[o][s] = A[o * 4 + 0] * xin[0][s] + A[o * 4 + 1] * xin[1][s] +
                        A[o * 4 + 2] * xin[2][s] + A[o * 4 + 3];
    }

    #pragma unroll
    for (int o = 0; o < 3; ++o) {
        float4 v = make_float4(res[o][0], res[o][1], res[o][2], res[o][3]);
        *(float4*)(out + ((size_t)b * 3 + o) * HW + pix0) = v;
    }
}

// ---------------------------------------------------------------------------
// Host-side orchestration
// ---------------------------------------------------------------------------
static inline int ilog2(int v) { int s = 0; while ((1 << s) < v) ++s; return s; }

static void launch_conv(hipStream_t stream, const float* in, const float* w,
                        const float* bias, const float* bns, const float* bnb,
                        float* out, int Cin, int Cout, int Hin, int Win,
                        int stride, int ksize, int pad, int relu)
{
    const int Hout = (Hin + 2 * pad - ksize) / stride + 1;
    const int Wout = (Win + 2 * pad - ksize) / stride + 1;
    dim3 grid((Hout * Wout + 63) / 64, (Cout + 15) / 16, 2);
    if (ksize == 3)
        conv_wmma_kernel<3><<<grid, 128, 0, stream>>>(in, w, bias, bns, bnb, out,
            Cin, Cout, Hin, Win, Hout, ilog2(Wout), stride, pad, relu);
    else
        conv_wmma_kernel<1><<<grid, 128, 0, stream>>>(in, w, bias, bns, bnb, out,
            Cin, Cout, Hin, Win, Hout, ilog2(Wout), stride, pad, relu);
}

extern "C" void kernel_launch(void* const* d_in, const int* in_sizes, int n_in,
                              void* d_out, int out_size, void* d_ws, size_t ws_size,
                              hipStream_t stream) {
    (void)in_sizes; (void)n_in; (void)out_size; (void)ws_size;

    // --- inputs, in setup_inputs() dict order (lists expanded in order) ---
    const float* image_lowres  = (const float*)d_in[0];   // (2,3,256,256)
    const float* image_fullres = (const float*)d_in[1];   // (2,3,1024,1024)
    const float* splat_w[4] = {(const float*)d_in[2], (const float*)d_in[3],
                               (const float*)d_in[4], (const float*)d_in[5]};
    const float* splat_b[4] = {(const float*)d_in[6], (const float*)d_in[7],
                               (const float*)d_in[8], (const float*)d_in[9]};
    const float* splat_bn_s[3] = {(const float*)d_in[10], (const float*)d_in[11],
                                  (const float*)d_in[12]};
    const float* splat_bn_b[3] = {(const float*)d_in[13], (const float*)d_in[14],
                                  (const float*)d_in[15]};
    const float* glob_w[2]    = {(const float*)d_in[16], (const float*)d_in[17]};
    const float* glob_b[2]    = {(const float*)d_in[18], (const float*)d_in[19]};
    const float* glob_bn_s[2] = {(const float*)d_in[20], (const float*)d_in[21]};
    const float* glob_bn_b[2] = {(const float*)d_in[22], (const float*)d_in[23]};
    const float* fc_w[3] = {(const float*)d_in[24], (const float*)d_in[25],
                            (const float*)d_in[26]};
    const float* fc_b[3] = {(const float*)d_in[27], (const float*)d_in[28],
                            (const float*)d_in[29]};
    const float* loc_w0    = (const float*)d_in[30];
    const float* loc_b0    = (const float*)d_in[31];
    const float* loc_bn0_s = (const float*)d_in[32];
    const float* loc_bn0_b = (const float*)d_in[33];
    const float* loc_w1    = (const float*)d_in[34];
    const float* loc_bn1_s = (const float*)d_in[35];
    const float* loc_bn1_b = (const float*)d_in[36];
    const float* pred_w    = (const float*)d_in[37];
    const float* pred_b    = (const float*)d_in[38];
    const float* pred_bn_s = (const float*)d_in[39];
    const float* pred_bn_b = (const float*)d_in[40];
    const float* ccm_w     = (const float*)d_in[41];
    const float* ccm_b     = (const float*)d_in[42];
    const float* shifts    = (const float*)d_in[43];
    const float* slopes    = (const float*)d_in[44];
    const float* proj_w    = (const float*)d_in[45];
    const float* proj_b    = (const float*)d_in[46];

    float* out = (float*)d_out;

    // --- workspace layout (floats) ---
    float* ws = (float*)d_ws;
    size_t off = 0;
    float* a1   = ws + off; off += (size_t)2 * 8  * 128 * 128;
    float* a2   = ws + off; off += (size_t)2 * 16 * 64  * 64;
    float* a3   = ws + off; off += (size_t)2 * 32 * 32  * 32;
    float* a4   = ws + off; off += (size_t)2 * 64 * 16  * 16;   // splat features
    float* g1   = ws + off; off += (size_t)2 * 64 * 8   * 8;
    float* g2   = ws + off; off += (size_t)2 * 64 * 4   * 4;
    float* fc1  = ws + off; off += (size_t)2 * 256;
    float* fc2  = ws + off; off += (size_t)2 * 128;
    float* fc3  = ws + off; off += (size_t)2 * 64;
    float* l1   = ws + off; off += (size_t)2 * 64 * 16 * 16;
    float* l2   = ws + off; off += (size_t)2 * 64 * 16 * 16;
    float* fuse = ws + off; off += (size_t)2 * 64 * 16 * 16;
    float* pred = ws + off; off += (size_t)2 * 96 * 16 * 16;    // coeff grid

    // --- splat path (stride-2 3x3 convs) ---
    launch_conv(stream, image_lowres, splat_w[0], splat_b[0], nullptr, nullptr,
                a1, 3, 8, 256, 256, 2, 3, 1, 1);
    launch_conv(stream, a1, splat_w[1], splat_b[1], splat_bn_s[0], splat_bn_b[0],
                a2, 8, 16, 128, 128, 2, 3, 1, 1);
    launch_conv(stream, a2, splat_w[2], splat_b[2], splat_bn_s[1], splat_bn_b[1],
                a3, 16, 32, 64, 64, 2, 3, 1, 1);
    launch_conv(stream, a3, splat_w[3], splat_b[3], splat_bn_s[2], splat_bn_b[2],
                a4, 32, 64, 32, 32, 2, 3, 1, 1);

    // --- global path: 2 stride-2 convs + 3 FC (async-LDS staged) ---
    launch_conv(stream, a4, glob_w[0], glob_b[0], glob_bn_s[0], glob_bn_b[0],
                g1, 64, 64, 16, 16, 2, 3, 1, 1);
    launch_conv(stream, g1, glob_w[1], glob_b[1], glob_bn_s[1], glob_bn_b[1],
                g2, 64, 64, 8, 8, 2, 3, 1, 1);
    fc_kernel<<<1, 256, 0, stream>>>(g2,  fc_w[0], fc_b[0], fc1, 1024, 256, 1);
    fc_kernel<<<1, 256, 0, stream>>>(fc1, fc_w[1], fc_b[1], fc2, 256, 128, 1);
    fc_kernel<<<1, 256, 0, stream>>>(fc2, fc_w[2], fc_b[2], fc3, 128, 64, 0);

    // --- local path: 2 stride-1 3x3 convs ---
    launch_conv(stream, a4, loc_w0, loc_b0, loc_bn0_s, loc_bn0_b,
                l1, 64, 64, 16, 16, 1, 3, 1, 1);
    launch_conv(stream, l1, loc_w1, nullptr, loc_bn1_s, loc_bn1_b,
                l2, 64, 64, 16, 16, 1, 3, 1, 0);

    // --- fusion + 1x1 prediction conv (bilateral grid coefficients) ---
    {
        const int total = 2 * 64 * 256;
        fuse_kernel<<<dim3((total + 255) / 256), 256, 0, stream>>>(l2, fc3, fuse, 64, 256, total);
    }
    launch_conv(stream, fuse, pred_w, pred_b, pred_bn_s, pred_bn_b,
                pred, 64, 96, 16, 16, 1, 1, 0, 0);

    // --- fused guide + slice + apply at full resolution (4 px / thread) ---
    {
        const int H = 1024, W = 1024;
        dim3 grid(((H * W / 4) + 255) / 256, 1, 2);
        guide_slice_kernel<<<grid, 256, 0, stream>>>(
            image_fullres, pred, ccm_w, ccm_b, shifts, slopes, proj_w, proj_b,
            out, H, W);
    }
}
